// GNNBlock_6098853560867
// MI455X (gfx1250) — matrix operand order repaired and checked by reading, not compiled, not run on hardware
//
#include <hip/hip_runtime.h>
#include <hip/hip_bf16.h>

// ---------------------------------------------------------------------------
// GNN block (2x DenseGAT, fully-connected equal-size graphs) for gfx1250.
// B=512 graphs, N=128 nodes, D0=128, D1=D2=64, H=3 heads, lin-out = H*64 = 192.
// GEMMs on v_wmma_f32_16x16x32_f16. Layer-2 lin is fused into the layer-1
// aggregation kernel (LDS re-aliased after a barrier), attention scores are
// reduced in-register in the GEMM epilogues, and per-graph feature tiles are
// staged into LDS with global_load_async_to_lds_b128 (ASYNCcnt).
// ---------------------------------------------------------------------------

typedef __attribute__((ext_vector_type(16))) _Float16 v16h;
typedef __attribute__((ext_vector_type(8)))  _Float16 v8h;
typedef __attribute__((ext_vector_type(8)))  float    v8f;

#define NB        512
#define NN        128
#define LINOUT    192
#define CPH       64      // channels per head
#define NHEAD     3
#define ATT_SLOPE 0.2f
#define ACT_SLOPE 0.01f
#define HDS       (NN + 8)    // Hdt row stride (halves)
#define MIDS      (CPH + 8)   // mid / Wt2 row stride (halves)

static __device__ __forceinline__ float leaky(float v, float s) {
    return v > 0.0f ? v : s * v;
}

// load 16 contiguous halves (16B-aligned) from LDS into a v16h
static __device__ __forceinline__ v16h ld16h(const _Float16* p) {
    union { v16h v; v8h h[2]; } u;
    u.h[0] = *(const v8h*)(p);
    u.h[1] = *(const v8h*)(p + 8);
    return u.v;
}

// assemble a v16h A-fragment from two 8-half runs
static __device__ __forceinline__ v16h ld2x8h(const _Float16* lo, const _Float16* hi) {
    union { v16h v; v8h h[2]; } u;
    u.h[0] = *(const v8h*)(lo);
    u.h[1] = *(const v8h*)(hi);
    return u.v;
}

// ---------------------------------------------------------------------------
// Kernel 1: h = X[row, :128] @ W1[128, 192]
//   - writes h TRANSPOSED per graph: HT[b][c][j] (f16), packed b128 stores
//   - fuses per-node attention scores: S/D[row, head] = <h_row_head, att>
// 4 waves / block, each wave computes a 16x192 strip. W transposed in LDS.
// ---------------------------------------------------------------------------
__global__ __launch_bounds__(128) void lin1_kernel(const float* __restrict__ X,
                                                   const float* __restrict__ W,
                                                   const float* __restrict__ asrc,
                                                   const float* __restrict__ adst,
                                                   _Float16* __restrict__ HT,
                                                   float* __restrict__ S,
                                                   float* __restrict__ D) {
    constexpr int DIN = 128;
    __shared__ __attribute__((aligned(16))) _Float16 Wt[LINOUT][DIN + 8]; // Wt[c][k]
    const int tid = threadIdx.x;

    for (int id = tid; id < LINOUT * DIN; id += 128) {
        int c = id / DIN, k = id - c * DIN;
        Wt[c][k] = (_Float16)W[k * LINOUT + c];
    }
    __syncthreads();

    const int wave = tid >> 5;
    const int lane = tid & 31;
    const int half = lane >> 4;
    const int ln   = lane & 15;
    const long rowBase = (long)blockIdx.x * 64 + wave * 16;
    const float* xrow = X + (rowBase + ln) * DIN;

    float asv[12], adv[12];
#pragma unroll
    for (int nt = 0; nt < 12; ++nt) {
        asv[nt] = asrc[nt * 16 + ln];
        adv[nt] = adst[nt * 16 + ln];
    }

    v8f acc[12] = {};

#pragma unroll
    for (int kk = 0; kk < DIN / 32; ++kk) {
        const int k0 = kk * 32;
        v16h a;
#pragma unroll
        for (int e = 0; e < 8; ++e)
            a[e] = (_Float16)xrow[k0 + 8 * half + e];
#pragma unroll
        for (int e = 0; e < 8; ++e)
            a[8 + e] = (_Float16)xrow[k0 + 16 + 8 * half + e];

#pragma unroll
        for (int nt = 0; nt < 12; ++nt) {
            const _Float16* bp = &Wt[nt * 16 + ln][k0];
            v16h b = ld2x8h(bp + 8 * half, bp + 16 + 8 * half);
            acc[nt] = __builtin_amdgcn_wmma_f32_16x16x32_f16(
                false, a, false, b, (short)0, acc[nt], false, false);
        }
    }

    // ---- packed transposed store: 8 consecutive j per lane -> b128 ----
    const int bIdx = (int)(rowBase >> 7);
    const int j0   = (int)(rowBase & (NN - 1)) + 8 * half;
    _Float16* ht = HT + (long)bIdx * LINOUT * NN;
#pragma unroll
    for (int nt = 0; nt < 12; ++nt) {
        v8h hv;
#pragma unroll
        for (int r = 0; r < 8; ++r) hv[r] = (_Float16)acc[nt][r];
        *(v8h*)(ht + (long)(nt * 16 + ln) * NN + j0) = hv;
    }

    // ---- fused per-node scores (f32, from accumulators) ----
#pragma unroll
    for (int h = 0; h < NHEAD; ++h) {
#pragma unroll
        for (int r = 0; r < 8; ++r) {
            float ps = 0.0f, pd = 0.0f;
#pragma unroll
            for (int q = 0; q < 4; ++q) {
                const int nt = h * 4 + q;
                const float hv = acc[nt][r];
                ps += hv * asv[nt];
                pd += hv * adv[nt];
            }
#pragma unroll
            for (int off = 1; off < 16; off <<= 1) {   // reduce over ln (stays in half)
                ps += __shfl_xor(ps, off, 32);
                pd += __shfl_xor(pd, off, 32);
            }
            if (ln == 0) {
                long row = rowBase + r + 8 * half;
                S[row * NHEAD + h] = ps;
                D[row * NHEAD + h] = pd;
            }
        }
    }
}

// ---------------------------------------------------------------------------
// Kernel 2: fused layer-1 aggregation + layer-2 lin, one workgroup per graph.
// Phase 1: attention aggregation (async-LDS staged B operand, exp A-fragments
//          built in registers, 1/rowsum folded in), bias + leaky(0.01) -> mid.
// Phase 2: LDS re-aliased: mid[128][72] f16 + Wt2[192][72] f16; GEMM
//          mid @ W2 -> HT2 (packed transposed b128 stores) + layer-2 scores.
// ---------------------------------------------------------------------------
__global__ __launch_bounds__(256) void agg1_lin2_kernel(const _Float16* __restrict__ HT1,
                                                        const float* __restrict__ S1,
                                                        const float* __restrict__ D1,
                                                        const float* __restrict__ bias1,
                                                        const float* __restrict__ W2,
                                                        const float* __restrict__ asrc2,
                                                        const float* __restrict__ adst2,
                                                        _Float16* __restrict__ HT2,
                                                        float* __restrict__ S2,
                                                        float* __restrict__ D2) {
    // phase 1: Hdt[192][136] f16 (52,224 B)
    // phase 2: mid[128][72] f16 (18,432 B) + Wt2[192][72] f16 (27,648 B)
    __shared__ __attribute__((aligned(16))) char smem[LINOUT * HDS * 2];
    __shared__ float sh_s[NN * NHEAD];

    _Float16* Hdt = (_Float16*)smem;                                 // [c][j], stride HDS
    _Float16* mid = (_Float16*)smem;                                 // [row][c], stride MIDS
    _Float16* Wt2 = (_Float16*)(smem + NN * MIDS * 2);               // [c][k], stride MIDS

    const int b   = blockIdx.x;
    const int tid = threadIdx.x;

    // ---- async copy HT1 graph block (contiguous 48 KB) into padded LDS ----
    {
        const _Float16* srcT = HT1 + (long)b * LINOUT * NN;
        const unsigned ldsBase = (unsigned)(uintptr_t)Hdt;
        for (int id = tid; id < (LINOUT * NN) / 8; id += 256) {      // 16B chunks
            const int row = id >> 4;
            const int cir = id & 15;
            const unsigned lds = ldsBase + (unsigned)(row * HDS * 2 + cir * 16);
            const _Float16* g = srcT + (long)id * 8;
            asm volatile("global_load_async_to_lds_b128 %0, %1, off"
                         :: "v"(lds), "v"(g) : "memory");
        }
    }
    for (int id = tid; id < NN * NHEAD; id += 256)
        sh_s[id] = S1[b * NN * NHEAD + id];
    asm volatile("s_wait_asynccnt 0" ::: "memory");
    __syncthreads();

    const int wave = tid >> 5;
    const int lane = tid & 31;
    const int half = lane >> 4;
    const int ln   = lane & 15;
    const int m0   = wave * 16;
    const int i    = m0 + ln;                  // this lane's attention row

    v8f tot[4] = {};

#pragma unroll
    for (int head = 0; head < NHEAD; ++head) {
        const float d_i = D1[((long)b * NN + i) * NHEAD + head];

        float smax = -1e30f;
        for (int j = 0; j < NN; ++j)
            smax = fmaxf(smax, sh_s[j * NHEAD + head]);
        const float rm = leaky(smax + d_i, ATT_SLOPE);   // leaky is monotone

        float sum = 0.0f;
        for (int j = half * 64; j < half * 64 + 64; ++j)
            sum += __expf(leaky(sh_s[j * NHEAD + head] + d_i, ATT_SLOPE) - rm);
        sum += __shfl_xor(sum, 16, 32);
        const float inv = 1.0f / sum;

        v8f acc[4] = {};
#pragma unroll
        for (int kk = 0; kk < 4; ++kk) {
            const int k0 = kk * 32;
            v16h a;
#pragma unroll
            for (int e = 0; e < 8; ++e) {
                int k = k0 + 8 * half + e;
                a[e] = (_Float16)__expf(
                    leaky(sh_s[k * NHEAD + head] + d_i, ATT_SLOPE) - rm);
            }
#pragma unroll
            for (int e = 0; e < 8; ++e) {
                int k = k0 + 16 + 8 * half + e;
                a[8 + e] = (_Float16)__expf(
                    leaky(sh_s[k * NHEAD + head] + d_i, ATT_SLOPE) - rm);
            }
#pragma unroll
            for (int nt = 0; nt < 4; ++nt) {
                const _Float16* bp = &Hdt[(head * CPH + nt * 16 + ln) * HDS + k0];
                v16h bf = ld2x8h(bp + 8 * half, bp + 16 + 8 * half);
                acc[nt] = __builtin_amdgcn_wmma_f32_16x16x32_f16(
                    false, a, false, bf, (short)0, acc[nt], false, false);
            }
        }

        float invr[8];
#pragma unroll
        for (int r = 0; r < 8; ++r)
            invr[r] = __shfl(inv, r + 8 * half, 32);
#pragma unroll
        for (int nt = 0; nt < 4; ++nt)
#pragma unroll
            for (int r = 0; r < 8; ++r)
                tot[nt][r] += acc[nt][r] * invr[r];
    }

    __syncthreads();   // everyone done reading Hdt; LDS re-aliased below

    // ---- mid = leaky(tot/3 + b1) -> LDS (f16, row-major) ----
    const float inv3 = 1.0f / 3.0f;
#pragma unroll
    for (int nt = 0; nt < 4; ++nt) {
        const int c = nt * 16 + ln;
        const float bb = bias1[c];
#pragma unroll
        for (int r = 0; r < 8; ++r) {
            float v = tot[nt][r] * inv3 + bb;
            v = leaky(v, ACT_SLOPE);
            mid[(m0 + r + 8 * half) * MIDS + c] = (_Float16)v;
        }
    }
    // ---- load W2 transposed into LDS ----
    for (int id = tid; id < LINOUT * CPH; id += 256) {
        int c = id >> 6, k = id & 63;
        Wt2[c * MIDS + k] = (_Float16)W2[k * LINOUT + c];
    }
    __syncthreads();

    // ---- layer-2 lin GEMM: [128 x 64] @ [64 x 192], wave = 16-row strip ----
    float asv[12], adv[12];
#pragma unroll
    for (int nt = 0; nt < 12; ++nt) {
        asv[nt] = asrc2[nt * 16 + ln];
        adv[nt] = adst2[nt * 16 + ln];
    }

    v8f acc2[12] = {};
    const _Float16* arow = mid + (m0 + ln) * MIDS;
#pragma unroll
    for (int kk = 0; kk < 2; ++kk) {
        const int k0 = kk * 32;
        v16h a = ld2x8h(arow + k0 + 8 * half, arow + k0 + 16 + 8 * half);
#pragma unroll
        for (int nt = 0; nt < 12; ++nt) {
            const _Float16* bp = Wt2 + (nt * 16 + ln) * MIDS + k0;
            v16h bf = ld2x8h(bp + 8 * half, bp + 16 + 8 * half);
            acc2[nt] = __builtin_amdgcn_wmma_f32_16x16x32_f16(
                false, a, false, bf, (short)0, acc2[nt], false, false);
        }
    }

    // ---- packed transposed HT2 store ----
    const int j0 = m0 + 8 * half;
    _Float16* ht2 = HT2 + (long)b * LINOUT * NN;
#pragma unroll
    for (int nt = 0; nt < 12; ++nt) {
        v8h hv;
#pragma unroll
        for (int r = 0; r < 8; ++r) hv[r] = (_Float16)acc2[nt][r];
        *(v8h*)(ht2 + (long)(nt * 16 + ln) * NN + j0) = hv;
    }

    // ---- fused layer-2 scores ----
#pragma unroll
    for (int h = 0; h < NHEAD; ++h) {
#pragma unroll
        for (int r = 0; r < 8; ++r) {
            float ps = 0.0f, pd = 0.0f;
#pragma unroll
            for (int q = 0; q < 4; ++q) {
                const int nt = h * 4 + q;
                const float hv = acc2[nt][r];
                ps += hv * asv[nt];
                pd += hv * adv[nt];
            }
#pragma unroll
            for (int off = 1; off < 16; off <<= 1) {
                ps += __shfl_xor(ps, off, 32);
                pd += __shfl_xor(pd, off, 32);
            }
            if (ln == 0) {
                long row = (long)b * NN + m0 + r + 8 * half;
                S2[row * NHEAD + h] = ps;
                D2[row * NHEAD + h] = pd;
            }
        }
    }
}

// ---------------------------------------------------------------------------
// Kernel 3: layer-2 aggregation + bias + leaky + sum-pool -> [B, 64].
// ---------------------------------------------------------------------------
__global__ __launch_bounds__(256) void agg2_kernel(const _Float16* __restrict__ HT,
                                                   const float* __restrict__ S,
                                                   const float* __restrict__ Dsc,
                                                   const float* __restrict__ bias,
                                                   float* __restrict__ Out) {
    __shared__ __attribute__((aligned(16))) _Float16 Hdt[LINOUT][HDS]; // [c][j]
    __shared__ float sh_s[NN * NHEAD];
    __shared__ float outAcc[CPH];

    const int b   = blockIdx.x;
    const int tid = threadIdx.x;

    {
        const _Float16* srcT = HT + (long)b * LINOUT * NN;
        const unsigned ldsBase = (unsigned)(uintptr_t)&Hdt[0][0];
        for (int id = tid; id < (LINOUT * NN) / 8; id += 256) {
            const int row = id >> 4;
            const int cir = id & 15;
            const unsigned lds = ldsBase + (unsigned)(row * HDS * 2 + cir * 16);
            const _Float16* g = srcT + (long)id * 8;
            asm volatile("global_load_async_to_lds_b128 %0, %1, off"
                         :: "v"(lds), "v"(g) : "memory");
        }
    }
    for (int id = tid; id < NN * NHEAD; id += 256)
        sh_s[id] = S[b * NN * NHEAD + id];
    if (tid < CPH) outAcc[tid] = 0.0f;
    asm volatile("s_wait_asynccnt 0" ::: "memory");
    __syncthreads();

    const int wave = tid >> 5;
    const int lane = tid & 31;
    const int half = lane >> 4;
    const int ln   = lane & 15;
    const int m0   = wave * 16;
    const int i    = m0 + ln;

    v8f tot[4] = {};

#pragma unroll
    for (int head = 0; head < NHEAD; ++head) {
        const float d_i = Dsc[((long)b * NN + i) * NHEAD + head];

        float smax = -1e30f;
        for (int j = 0; j < NN; ++j)
            smax = fmaxf(smax, sh_s[j * NHEAD + head]);
        const float rm = leaky(smax + d_i, ATT_SLOPE);

        float sum = 0.0f;
        for (int j = half * 64; j < half * 64 + 64; ++j)
            sum += __expf(leaky(sh_s[j * NHEAD + head] + d_i, ATT_SLOPE) - rm);
        sum += __shfl_xor(sum, 16, 32);
        const float inv = 1.0f / sum;

        v8f acc[4] = {};
#pragma unroll
        for (int kk = 0; kk < 4; ++kk) {
            const int k0 = kk * 32;
            v16h a;
#pragma unroll
            for (int e = 0; e < 8; ++e) {
                int k = k0 + 8 * half + e;
                a[e] = (_Float16)__expf(
                    leaky(sh_s[k * NHEAD + head] + d_i, ATT_SLOPE) - rm);
            }
#pragma unroll
            for (int e = 0; e < 8; ++e) {
                int k = k0 + 16 + 8 * half + e;
                a[8 + e] = (_Float16)__expf(
                    leaky(sh_s[k * NHEAD + head] + d_i, ATT_SLOPE) - rm);
            }
#pragma unroll
            for (int nt = 0; nt < 4; ++nt) {
                const _Float16* bp = &Hdt[head * CPH + nt * 16 + ln][k0];
                v16h bf = ld2x8h(bp + 8 * half, bp + 16 + 8 * half);
                acc[nt] = __builtin_amdgcn_wmma_f32_16x16x32_f16(
                    false, a, false, bf, (short)0, acc[nt], false, false);
            }
        }

        float invr[8];
#pragma unroll
        for (int r = 0; r < 8; ++r)
            invr[r] = __shfl(inv, r + 8 * half, 32);
#pragma unroll
        for (int nt = 0; nt < 4; ++nt)
#pragma unroll
            for (int r = 0; r < 8; ++r)
                tot[nt][r] += acc[nt][r] * invr[r];
    }

    const float inv3 = 1.0f / 3.0f;
#pragma unroll
    for (int nt = 0; nt < 4; ++nt) {
        const int c = nt * 16 + ln;
        const float bb = bias[c];
        float part = 0.0f;
#pragma unroll
        for (int r = 0; r < 8; ++r) {
            float v = tot[nt][r] * inv3 + bb;
            part += leaky(v, ACT_SLOPE);
        }
        part += __shfl_xor(part, 16, 32);       // combine the two 8-row halves
        if (half == 0) atomicAdd(&outAcc[c], part);
    }
    __syncthreads();
    if (tid < CPH) Out[b * CPH + tid] = outAcc[tid];
}

// ---------------------------------------------------------------------------
// Launch sequence (stream-ordered): lin1 -> agg1+lin2 -> agg2
// ---------------------------------------------------------------------------
extern "C" void kernel_launch(void* const* d_in, const int* in_sizes, int n_in,
                              void* d_out, int out_size, void* d_ws, size_t ws_size,
                              hipStream_t stream) {
    (void)in_sizes; (void)n_in; (void)out_size; (void)ws_size;

    const float* x   = (const float*)d_in[0];
    const float* W1  = (const float*)d_in[2];
    const float* as1 = (const float*)d_in[3];
    const float* ad1 = (const float*)d_in[4];
    const float* b1  = (const float*)d_in[5];
    const float* W2  = (const float*)d_in[6];
    const float* as2 = (const float*)d_in[7];
    const float* ad2 = (const float*)d_in[8];
    const float* b2  = (const float*)d_in[9];
    float* out = (float*)d_out;

    char* ws = (char*)d_ws;
    size_t off = 0;
    _Float16* hT1 = (_Float16*)(ws + off); off += (size_t)NB * NN * LINOUT * 2;  // 24 MB
    _Float16* hT2 = (_Float16*)(ws + off); off += (size_t)NB * NN * LINOUT * 2;  // 24 MB
    float*    s1  = (float*)(ws + off);    off += (size_t)NB * NN * NHEAD * 4;
    float*    d1  = (float*)(ws + off);    off += (size_t)NB * NN * NHEAD * 4;
    float*    s2  = (float*)(ws + off);    off += (size_t)NB * NN * NHEAD * 4;
    float*    d2  = (float*)(ws + off);    off += (size_t)NB * NN * NHEAD * 4;

    lin1_kernel<<<(NB * NN) / 64, 128, 0, stream>>>(x, W1, as1, ad1, hT1, s1, d1);
    agg1_lin2_kernel<<<NB, 256, 0, stream>>>(hT1, s1, d1, b1, W2, as2, ad2, hT2, s2, d2);
    agg2_kernel<<<NB, 256, 0, stream>>>(hT2, s2, d2, b2, out);
}
